// NodeModel_ONE_10393820857011
// MI455X (gfx1250) — matrix operand order repaired and checked by reading, not compiled, not run on hardware
//
#include <hip/hip_runtime.h>
#include <hip/hip_bf16.h>

// ---------------------------------------------------------------------------
// NodeModel: out = concat(x, segment_sum(concat(x[row],edge_attr)@W1+b1, col)) @ W2 + b2
// N=100000 nodes (64 feat), E=1600000 edges (32 feat), HIDDEN=128.
// bf16 WMMA (v_wmma_f32_16x16x32_bf16) with f32 accumulation; fused scatter.
// Edge kernel: B register-resident per wave, A staged once through LDS,
// branch-free epilogue (E is a multiple of the 128-edge block tile).
// ---------------------------------------------------------------------------

#define NN      100000
#define NE      1600000
#define F_EDGE  32
#define F_NODE  64
#define HID     128
#define A_STRIDE 104   // padded bf16 stride per edge row in LDS (96 feats + pad)

static_assert(NE % 128 == 0, "edge kernel assumes no tail block");

typedef __attribute__((ext_vector_type(16))) __bf16 bf16x16;
typedef __attribute__((ext_vector_type(8)))  __bf16 bf16x8;
typedef __attribute__((ext_vector_type(8)))  float  f32x8;

__device__ __forceinline__ unsigned pack_bf16(float a, float b) {
    union { __bf16 h[2]; unsigned u; } u_;
    u_.h[0] = (__bf16)a; u_.h[1] = (__bf16)b;
    return u_.u;
}

// ---------------------------------------------------------------------------
// Kernel 0: zero the aggregation buffer (N x 128 f32) in workspace.
// ---------------------------------------------------------------------------
__global__ __launch_bounds__(256) void zero_agg(float* __restrict__ agg, long long n) {
    long long i = (long long)blockIdx.x * blockDim.x + threadIdx.x;
    long long stride = (long long)gridDim.x * blockDim.x;
    for (; i < n; i += stride) agg[i] = 0.0f;
}

// ---------------------------------------------------------------------------
// Kernel 1: per-edge MLP1 + scatter-add.
// Block = 256 threads = 8 waves, covers 128 edges x 128 outputs.
// Wave w: edge-half eh = w>>2 (64 edges), tile-pair tp = w&3 (32 outputs).
// B (W1 tiles) register-resident: 3 K-steps x 2 N-tiles = 48 VGPRs bf16.
// A staged once to LDS (bf16, padded rows); 4 subtiles x 3 ks x 2 WMMA = 24/wave.
// ---------------------------------------------------------------------------
__global__ __launch_bounds__(256) void edge_mlp_scatter(
    const float*     __restrict__ x,       // [N,64]
    const long long* __restrict__ eidx,    // [2,E] int64: row then col
    const float*     __restrict__ eattr,   // [E,32]
    const float*     __restrict__ W1,      // [96,128] row-major (k,n)
    const float*     __restrict__ b1,      // [128]
    float*           __restrict__ agg)     // [N,128] f32 accumulation target
{
    __shared__ __align__(16) __bf16 sA[128 * A_STRIDE];  // 128 edges x 96 feats (padded)
    __shared__ int sCol[128];

    const int tid = threadIdx.x;
    const long long ebase_blk = (long long)blockIdx.x * 128;

    // ---- Stage A: gather 128 edges' features (x[row] | edge_attr) -> LDS bf16.
    // Thread pair per edge: half 0 -> feats [0,48), half 1 -> feats [48,96).
    {
        const int e    = tid >> 1;
        const int half = tid & 1;
        const long long ge = ebase_blk + e;              // always < NE (no tail)
        const long long r  = eidx[ge];                   // edge_index[0][e]
        unsigned* dst = (unsigned*)&sA[e * A_STRIDE + half * 48];
        if (half == 0) {
            const float4* src = (const float4*)(x + r * F_NODE);      // feats 0..47
            __builtin_prefetch(eattr + (ebase_blk + 128) * F_EDGE + e * 16, 0, 1);
#pragma unroll
            for (int i = 0; i < 12; ++i) {
                const float4 f = src[i];
                dst[2 * i]     = pack_bf16(f.x, f.y);
                dst[2 * i + 1] = pack_bf16(f.z, f.w);
            }
        } else {
            const float4* s1 = (const float4*)(x + r * F_NODE + 48);  // feats 48..63
            const float4* s2 = (const float4*)(eattr + ge * F_EDGE);  // feats 64..95
#pragma unroll
            for (int i = 0; i < 4; ++i) {
                const float4 f = s1[i];
                dst[2 * i]     = pack_bf16(f.x, f.y);
                dst[2 * i + 1] = pack_bf16(f.z, f.w);
            }
#pragma unroll
            for (int i = 0; i < 8; ++i) {
                const float4 f = s2[i];
                dst[8 + 2 * i]     = pack_bf16(f.x, f.y);
                dst[8 + 2 * i + 1] = pack_bf16(f.z, f.w);
            }
        }
        if (tid < 128) {
            sCol[tid] = (int)eidx[(long long)NE + ebase_blk + tid];   // edge_index[1][e]
        }
    }

    const int wave = tid >> 5;
    const int lane = tid & 31;
    const int m    = lane & 15;   // row within subtile / col within N-tile
    const int kh   = lane >> 4;   // K-half selector per WMMA A/B layouts
    const int eh   = wave >> 2;   // edge half: 64 edges
    const int tp   = wave & 3;    // N-tile pair: columns [tp*32, tp*32+32)

    // ---- B register-resident: W1 tiles for this wave's 2 N-tiles, all 96 K.
    // B layout: lane = column, lanes 0-15 hold K 0..7,16..23, lanes 16-31 K 8..15,24..31.
    bf16x16 Breg[3][2];
    float   bias[2];
#pragma unroll
    for (int tt = 0; tt < 2; ++tt) {
        const int ncol = (tp * 2 + tt) * 16 + m;
        bias[tt] = b1[ncol];
#pragma unroll
        for (int ks = 0; ks < 3; ++ks) {
#pragma unroll
            for (int v = 0; v < 8; ++v) {
                const int k = ks * 32 + kh * 8 + ((v < 4) ? (v * 2) : (16 + (v - 4) * 2));
                Breg[ks][tt][2 * v]     = (__bf16)W1[k * HID + ncol];
                Breg[ks][tt][2 * v + 1] = (__bf16)W1[(k + 1) * HID + ncol];
            }
        }
    }

    __syncthreads();

    // ---- 4 subtiles of 16 edges each; inner loop has no B LDS traffic.
#pragma unroll
    for (int s = 0; s < 4; ++s) {
        const int elocal = eh * 64 + s * 16;            // subtile base within block

        f32x8 acc[2];
        acc[0] = (f32x8){0,0,0,0,0,0,0,0};
        acc[1] = (f32x8){0,0,0,0,0,0,0,0};

#pragma unroll
        for (int ks = 0; ks < 3; ++ks) {
            // A layout: lane row = m (+half select kh); K runs of 8 contiguous:
            // [ks*32 + kh*8, +8) and [ks*32 + 16 + kh*8, +8)  -> two b128 LDS loads.
            const __bf16* ap = &sA[(elocal + m) * A_STRIDE + ks * 32 + kh * 8];
            const bf16x8 lo = *(const bf16x8*)ap;
            const bf16x8 hi = *(const bf16x8*)(ap + 16);
            const bf16x16 a = __builtin_shufflevector(lo, hi,
                0, 1, 2, 3, 4, 5, 6, 7, 8, 9, 10, 11, 12, 13, 14, 15);

            acc[0] = __builtin_amdgcn_wmma_f32_16x16x32_bf16(
                false, a, false, Breg[ks][0], (short)0, acc[0], false, false);
            acc[1] = __builtin_amdgcn_wmma_f32_16x16x32_bf16(
                false, a, false, Breg[ks][1], (short)0, acc[1], false, false);
        }

        // ---- bias + scatter (branch-free: every edge in the block is valid).
        // D layout: lane holds col ncol; VGPR vr = row vr + kh*8.
#pragma unroll
        for (int tt = 0; tt < 2; ++tt) {
            const int ncol = (tp * 2 + tt) * 16 + m;
#pragma unroll
            for (int vr = 0; vr < 8; ++vr) {
                const int mr  = vr + kh * 8;
                const unsigned c = (unsigned)sCol[elocal + mr];
                atomicAdd(agg + ((c << 7) + (unsigned)ncol), acc[tt][vr] + bias[tt]);
            }
        }
    }
}

// ---------------------------------------------------------------------------
// Kernel 2: node MLP2: out = [x | agg] @ W2 + b2.
// Each wave: 16 nodes x 64 outputs; K=192 -> 6 K-steps x 4 N-tiles = 24 WMMA.
// ---------------------------------------------------------------------------
__global__ __launch_bounds__(256) void node_mlp(
    const float* __restrict__ x,     // [N,64]
    const float* __restrict__ agg,   // [N,128]
    const float* __restrict__ W2,    // [192,64] row-major (k,n)
    const float* __restrict__ b2,    // [64]
    float*       __restrict__ out)   // [N,64]
{
    __shared__ __bf16 sW2[F_NODE * 192];   // n-major: sW2[n*192 + k] (24 KB)
    __shared__ float  sB2[F_NODE];

    const int tid = threadIdx.x;
    for (int i = tid; i < F_NODE * 192; i += 256) {
        const int n = i / 192, k = i % 192;
        sW2[i] = (__bf16)W2[k * F_NODE + n];
    }
    if (tid < F_NODE) sB2[tid] = b2[tid];
    __syncthreads();

    const int wave = tid >> 5;
    const int lane = tid & 31;
    const int m    = lane & 15;
    const int kh   = lane >> 4;

    const long long nd_base = (long long)blockIdx.x * 128 + (long long)wave * 16;

    long long nd = nd_base + m;
    if (nd >= NN) nd = NN - 1;                      // clamp loads; stores guarded
    const float* xr = x   + nd * F_NODE;
    const float* ag = agg + nd * HID;

    f32x8 acc[4];
#pragma unroll
    for (int t = 0; t < 4; ++t) acc[t] = (f32x8){0,0,0,0,0,0,0,0};

#pragma unroll
    for (int ks = 0; ks < 6; ++ks) {
        bf16x16 a;
#pragma unroll
        for (int v = 0; v < 8; ++v) {
            const int kb = ks * 32 + kh * 16 + v * 2;
            const float f0 = (kb     < F_NODE) ? xr[kb]     : ag[kb - F_NODE];
            const float f1 = (kb + 1 < F_NODE) ? xr[kb + 1] : ag[kb + 1 - F_NODE];
            a[2 * v]     = (__bf16)f0;
            a[2 * v + 1] = (__bf16)f1;
        }
#pragma unroll
        for (int t = 0; t < 4; ++t) {
            const int ncol = t * 16 + m;
            const __bf16* wb = sW2 + ncol * 192 + ks * 32 + kh * 16;
            bf16x16 b;
#pragma unroll
            for (int v = 0; v < 8; ++v) {
                b[2 * v]     = wb[2 * v];
                b[2 * v + 1] = wb[2 * v + 1];
            }
            acc[t] = __builtin_amdgcn_wmma_f32_16x16x32_bf16(
                false, a, false, b, (short)0, acc[t], false, false);
        }
    }

#pragma unroll
    for (int t = 0; t < 4; ++t) {
        const int   ncol = t * 16 + m;
        const float bias = sB2[ncol];
#pragma unroll
        for (int vr = 0; vr < 8; ++vr) {
            const int       mr  = vr + kh * 8;
            const long long nde = nd_base + mr;
            if (nde < NN) out[nde * F_NODE + ncol] = acc[t][vr] + bias;
        }
    }
}

// ---------------------------------------------------------------------------
// Launcher
// Inputs (setup_inputs order): 0:x 1:edge_index(int64) 2:edge_attr 3:u 4:batch
//                              5:W1 6:b1 7:W2 8:b2.  u/batch unused by reference.
// d_out: [N,64] f32.  d_ws: agg [N,128] f32 (51.2 MB).
// ---------------------------------------------------------------------------
extern "C" void kernel_launch(void* const* d_in, const int* in_sizes, int n_in,
                              void* d_out, int out_size, void* d_ws, size_t ws_size,
                              hipStream_t stream) {
    const float*     x     = (const float*)d_in[0];
    const long long* eidx  = (const long long*)d_in[1];   // int64 per reference
    const float*     eattr = (const float*)d_in[2];
    const float*     W1    = (const float*)d_in[5];
    const float*     b1    = (const float*)d_in[6];
    const float*     W2    = (const float*)d_in[7];
    const float*     b2    = (const float*)d_in[8];
    float*           out   = (float*)d_out;
    float*           agg   = (float*)d_ws;                // N*128 f32

    const long long agg_elems = (long long)NN * HID;
    zero_agg<<<2048, 256, 0, stream>>>(agg, agg_elems);

    // 128 edges per block
    const int edge_blocks = (NE + 127) / 128;             // 12500
    edge_mlp_scatter<<<edge_blocks, 256, 0, stream>>>(x, eidx, eattr, W1, b1, agg);

    // 128 nodes per block
    const int node_blocks = (NN + 127) / 128;             // 782
    node_mlp<<<node_blocks, 256, 0, stream>>>(x, agg, W2, b2, out);
}
